// LuongAttnDecoderRNN_28449863369538
// MI455X (gfx1250) — matrix operand order repaired and checked by reading, not compiled, not run on hardware
//
#include <hip/hip_runtime.h>
#include <hip/hip_bf16.h>
#include <math.h>

// Problem dims
#define B_ 128
#define L_ 256
#define H_ 1024
#define W_ 512
#define V_ 50257
#define G_ 4096 /* 4*H */

typedef __attribute__((ext_vector_type(16))) __bf16 v16bf;
typedef __attribute__((ext_vector_type(8)))  __bf16 v8bf;
typedef __attribute__((ext_vector_type(4)))  __bf16 v4bf;
typedef __attribute__((ext_vector_type(8)))  float  v8f;
typedef __attribute__((ext_vector_type(2)))  float  v2f;
typedef __attribute__((ext_vector_type(4)))  int    v4i;

// ---------------------------------------------------------------------------
// CDNA5 async global->LDS staging (ASYNCcnt-tracked); fall back to
// load + ds_store if the builtin is absent on this toolchain.
// Builtin signature (from clang diagnostic): (v4i AS1* src, v4i AS3* dst,
// imm offset, imm cpol) — AS1 prints as __device__ in HIP diagnostics.
// ---------------------------------------------------------------------------
#if __has_builtin(__builtin_amdgcn_global_load_async_to_lds_b128)
#define HAVE_ASYNC_LDS 1
#else
#define HAVE_ASYNC_LDS 0
#endif

__device__ __forceinline__ void stage16(const __bf16* g, __bf16* l) {
#if HAVE_ASYNC_LDS
  __builtin_amdgcn_global_load_async_to_lds_b128(
      (__attribute__((address_space(1))) v4i*)g,
      (__attribute__((address_space(3))) v4i*)l, 0, 0);
#else
  *(uint4*)l = *(const uint4*)g;
#endif
}

#if HAVE_ASYNC_LDS
#if __has_builtin(__builtin_amdgcn_s_wait_asynccnt)
#define ASYNC_WAIT() __builtin_amdgcn_s_wait_asynccnt(0)
#else
#define ASYNC_WAIT() asm volatile("s_wait_asynccnt 0x0" ::: "memory")
#endif
#else
#define ASYNC_WAIT() ((void)0)
#endif

// ---------------------------------------------------------------------------
// bf16 WMMA fragment loaders (contiguous k = 16h + s in both A and B, wave32)
// A (MxK row-major): lane l -> row l&15; K-chunks {0..15} (lanes 0-15)
//                    or {16..31} (lanes 16-31) relative to k0.
// B from (N,K) row-major weight: lane l -> col n=l&15, 16 contiguous K values
//                    starting at k0 + (lane<16 ? 0 : 16)  => one 32B load.
// ---------------------------------------------------------------------------
__device__ __forceinline__ v16bf ldA(const __bf16* A, int lda, int k0, int lane) {
  int m  = lane & 15;
  int kb = k0 + ((lane & 16) ? 16 : 0);
  const __bf16* p = A + (size_t)m * lda + kb;
  v8bf lo = *(const v8bf*)p;
  v8bf hi = *(const v8bf*)(p + 8);
  return __builtin_shufflevector(lo, hi, 0,1,2,3,4,5,6,7,8,9,10,11,12,13,14,15);
}
__device__ __forceinline__ v16bf ldB(const __bf16* Bm, int ldb, int k0, int lane) {
  int n  = lane & 15;
  int kb = k0 + ((lane & 16) ? 16 : 0);
  return *(const v16bf*)(Bm + (size_t)n * ldb + kb);
}
// B fragment from LDS-staged [64][BSTRIDE] slice (padded stride: 16 rows hit
// disjoint bank groups for the two ds_load_b128's per lane).
#define BSTRIDE 40
__device__ __forceinline__ v16bf ldBs(const __bf16* Bbase, int tt, int lane) {
  int n  = lane & 15;
  int kb = (lane & 16) ? 16 : 0;
  const __bf16* p = Bbase + (size_t)(tt * 16 + n) * BSTRIDE + kb;
  v8bf lo = *(const v8bf*)p;
  v8bf hi = *(const v8bf*)(p + 8);
  return __builtin_shufflevector(lo, hi, 0,1,2,3,4,5,6,7,8,9,10,11,12,13,14,15);
}
#define WMMA_BF16(a, b, c) \
  __builtin_amdgcn_wmma_f32_16x16x32_bf16(false, (a), false, (b), (short)0, (c), false, false)

// ---------------------------------------------------------------------------
// Elementwise prep kernels
// ---------------------------------------------------------------------------
__global__ void k_cast_bf16(const float* __restrict__ src, __bf16* __restrict__ dst, int n) {
  int i = blockIdx.x * 256 + threadIdx.x;
  if (i < n) dst[i] = (__bf16)src[i];
}

__global__ void k_zero_f32(float* __restrict__ p, int n) {
  int i = blockIdx.x * 256 + threadIdx.x;
  if (i < n) p[i] = 0.0f;
}

// x = emb[input_seq]  -> bf16 (B,W)
__global__ void k_embed(const int* __restrict__ idx, const float* __restrict__ emb,
                        __bf16* __restrict__ xbf) {
  int i = blockIdx.x * 256 + threadIdx.x;
  if (i >= B_ * W_) return;
  int b = i / W_, w = i % W_;
  int t = idx[b];
  t = t < 0 ? 0 : (t >= V_ ? V_ - 1 : t);
  xbf[i] = (__bf16)emb[(size_t)t * W_ + w];
}

// (L,B,H) f32 -> (B,L,H) bf16 (67 MB, lives in L2 for the energy GEMM)
__global__ void k_enc_cast(const float* __restrict__ enc_in, __bf16* __restrict__ encbf) {
  long long i = (long long)blockIdx.x * 256 + threadIdx.x;
  if (i >= (long long)L_ * B_ * H_) return;
  int h = (int)(i % H_);
  long long r = i / H_;
  int b = (int)(r % B_);
  int l = (int)(r / B_);
  encbf[((size_t)b * L_ + l) * H_ + h] = (__bf16)enc_in[i];
}

// split W1_w (H, 2H) into Wh (:, :H) and We (:, H:)
__global__ void k_split_w1(const float* __restrict__ W1w,
                           __bf16* __restrict__ Wh, __bf16* __restrict__ We) {
  int i = blockIdx.x * 256 + threadIdx.x;
  if (i >= H_ * 2 * H_) return;
  int row = i / (2 * H_), col = i % (2 * H_);
  float x = W1w[i];
  if (col < H_) Wh[(size_t)row * H_ + col] = (__bf16)x;
  else          We[(size_t)row * H_ + (col - H_)] = (__bf16)x;
}

// ---------------------------------------------------------------------------
// gates = x@W_ih^T + h_prev@W_hh^T + b_ih + b_hh   (128 x 4096, bf16 WMMA)
// block = 256 thr = 8 waves; wave w -> M-tile w; blockIdx.x -> N-tile (256)
// ---------------------------------------------------------------------------
__global__ void k_gates_gemm(const __bf16* __restrict__ xbf, const __bf16* __restrict__ hbf,
                             const __bf16* __restrict__ Wih, const __bf16* __restrict__ Whh,
                             const float* __restrict__ bih, const float* __restrict__ bhh,
                             float* __restrict__ gates) {
  int lane = threadIdx.x & 31;
  int m0   = (threadIdx.x >> 5) * 16;
  int n0   = blockIdx.x * 16;
  v8f acc = {};
  {
    const __bf16* A  = xbf + (size_t)m0 * W_;
    const __bf16* Bm = Wih + (size_t)n0 * W_;
    for (int k0 = 0; k0 < W_; k0 += 32)
      acc = WMMA_BF16(ldA(A, W_, k0, lane), ldB(Bm, W_, k0, lane), acc);
  }
  {
    const __bf16* A  = hbf + (size_t)m0 * H_;
    const __bf16* Bm = Whh + (size_t)n0 * H_;
    for (int k0 = 0; k0 < H_; k0 += 32)
      acc = WMMA_BF16(ldA(A, H_, k0, lane), ldB(Bm, H_, k0, lane), acc);
  }
  int col  = lane & 15;
  int half = (lane & 16) ? 8 : 0;
  float bias = bih[n0 + col] + bhh[n0 + col];
#pragma unroll
  for (int r = 0; r < 8; r++)
    gates[(size_t)(m0 + half + r) * G_ + n0 + col] = acc[r] + bias;
}

// LSTM pointwise: c = sig(f)*c0 + sig(i)*tanh(g); h = sig(o)*tanh(c)
__global__ void k_lstm_pw(const float* __restrict__ gates, const float* __restrict__ c0,
                          float* __restrict__ h_out, float* __restrict__ c_out,
                          __bf16* __restrict__ hbf, __bf16* __restrict__ concat_bf) {
  int i = blockIdx.x * 256 + threadIdx.x;
  if (i >= B_ * H_) return;
  int b = i / H_, h = i % H_;
  const float* g = gates + (size_t)b * G_;
  float gi = g[h], gf = g[H_ + h], gg = g[2 * H_ + h], go = g[3 * H_ + h];
  float si = 1.0f / (1.0f + __expf(-gi));
  float sf = 1.0f / (1.0f + __expf(-gf));
  float so = 1.0f / (1.0f + __expf(-go));
  float c  = sf * c0[i] + si * tanhf(gg);
  float hn = so * tanhf(c);
  c_out[i] = c;
  h_out[i] = hn;
  hbf[i] = (__bf16)hn;
  concat_bf[(size_t)b * (2 * H_) + h] = (__bf16)hn;
}

// hWh = h @ Wh^T + W1_b   (128 x 1024)
__global__ void k_hwh_gemm(const __bf16* __restrict__ hbf, const __bf16* __restrict__ Wh,
                           const float* __restrict__ b1, float* __restrict__ hWh) {
  int lane = threadIdx.x & 31;
  int m0   = (threadIdx.x >> 5) * 16;
  int n0   = blockIdx.x * 16;
  v8f acc = {};
  const __bf16* A  = hbf + (size_t)m0 * H_;
  const __bf16* Bm = Wh  + (size_t)n0 * H_;
  for (int k0 = 0; k0 < H_; k0 += 32)
    acc = WMMA_BF16(ldA(A, H_, k0, lane), ldB(Bm, H_, k0, lane), acc);
  int col  = lane & 15;
  int half = (lane & 16) ? 8 : 0;
  float bias = b1[n0 + col];
#pragma unroll
  for (int r = 0; r < 8; r++)
    hWh[(size_t)(m0 + half + r) * H_ + n0 + col] = acc[r] + bias;
}

// ---------------------------------------------------------------------------
// Energy + scores, fused. A 16(l) x 64(h') tile of enc@We^T per wave; the
// 64x32 bf16 B-slice is staged once per BLOCK into LDS with async DMA
// (double-buffered, overlapped with WMMA) instead of 8 waves re-loading it
// from L2. Epilogue: tanh(.+hWh)*v, cross-lane reduce, atomicAdd to scores.
// grid = (BL/16/8, H/64), block = 256 (8 waves)
// ---------------------------------------------------------------------------
__global__ void k_energy(const __bf16* __restrict__ encbf, const __bf16* __restrict__ We,
                         const float* __restrict__ hWh, const float* __restrict__ vvec,
                         float* __restrict__ scores) {
  __shared__ __bf16 Bs[2][64 * BSTRIDE];
  int tid  = threadIdx.x;
  int lane = tid & 31;
  int mt   = blockIdx.x * 8 + (tid >> 5);           // 0..2047 row tiles
  int b    = mt >> 4;                               // L=256 -> 16 tiles per batch
  int l0   = (mt & 15) * 16;
  int n0   = blockIdx.y * 64;
  const __bf16* A = encbf + ((size_t)b * L_ + l0) * H_;

  // staging coords: thread t moves one 16B chunk (8 bf16) per k-slice
  int srow = tid >> 2;                              // 0..63
  int scol = (tid & 3) * 8;                         // 0,8,16,24
  const __bf16* gB = We + (size_t)(n0 + srow) * H_ + scol;
  int loff = srow * BSTRIDE + scol;

  stage16(gB, &Bs[0][loff]);
  ASYNC_WAIT();
  __syncthreads();

  v8f a0 = {}, a1 = {}, a2 = {}, a3 = {};
  int buf = 0;
  for (int k0 = 0; k0 < H_; k0 += 32) {
    if (k0 + 32 < H_) stage16(gB + k0 + 32, &Bs[buf ^ 1][loff]);
    v16bf af = ldA(A, H_, k0, lane);
    const __bf16* Bbase = &Bs[buf][0];
    a0 = WMMA_BF16(af, ldBs(Bbase, 0, lane), a0);
    a1 = WMMA_BF16(af, ldBs(Bbase, 1, lane), a1);
    a2 = WMMA_BF16(af, ldBs(Bbase, 2, lane), a2);
    a3 = WMMA_BF16(af, ldBs(Bbase, 3, lane), a3);
    ASYNC_WAIT();
    __syncthreads();
    buf ^= 1;
  }

  int col  = lane & 15;
  int half = (lane & 16) ? 8 : 0;
  const float* hb = hWh + (size_t)b * H_;
  float h0 = hb[n0 + col],      v0 = vvec[n0 + col];
  float h1 = hb[n0 + 16 + col], v1 = vvec[n0 + 16 + col];
  float h2 = hb[n0 + 32 + col], v2 = vvec[n0 + 32 + col];
  float h3 = hb[n0 + 48 + col], v3 = vvec[n0 + 48 + col];
#pragma unroll
  for (int r = 0; r < 8; r++) {
    float s = tanhf(a0[r] + h0) * v0 + tanhf(a1[r] + h1) * v1 +
              tanhf(a2[r] + h2) * v2 + tanhf(a3[r] + h3) * v3;
    for (int m = 1; m < 16; m <<= 1) s += __shfl_xor(s, m, 32);
    if (col == r) atomicAdd(&scores[(size_t)b * L_ + l0 + half + r], s);
  }
}

// softmax over L, context = a @ enc, writes bf16 into concat[:, H:2H]
__global__ void k_softmax_ctx(const float* __restrict__ scores,
                              const __bf16* __restrict__ encbf,
                              __bf16* __restrict__ concat_bf) {
  __shared__ float sm[L_];
  __shared__ float red[L_];
  int b = blockIdx.x, t = threadIdx.x;
  float sv = scores[(size_t)b * L_ + t];
  red[t] = sv; __syncthreads();
  for (int off = 128; off > 0; off >>= 1) {
    if (t < off) red[t] = fmaxf(red[t], red[t + off]);
    __syncthreads();
  }
  float mx = red[0]; __syncthreads();
  float e = __expf(sv - mx);
  red[t] = e; __syncthreads();
  for (int off = 128; off > 0; off >>= 1) {
    if (t < off) red[t] += red[t + off];
    __syncthreads();
  }
  float inv = 1.0f / red[0];
  sm[t] = e * inv; __syncthreads();
  const __bf16* E = encbf + (size_t)b * L_ * H_;
  int h0 = t * 4;                              // H = 256 threads * 4
  float c0a = 0.f, c1 = 0.f, c2 = 0.f, c3 = 0.f;
  for (int l = 0; l < L_; l++) {
    v4bf e4 = *(const v4bf*)(E + (size_t)l * H_ + h0);
    float w = sm[l];
    c0a += w * (float)e4[0]; c1 += w * (float)e4[1];
    c2  += w * (float)e4[2]; c3 += w * (float)e4[3];
  }
  v4bf o = {(__bf16)c0a, (__bf16)c1, (__bf16)c2, (__bf16)c3};
  *(v4bf*)(concat_bf + (size_t)b * (2 * H_) + H_ + h0) = o;
}

// co = tanh(concat @ concat_w^T + concat_b)   (128 x 1024, K = 2048)
__global__ void k_concat_gemm(const __bf16* __restrict__ cat, const __bf16* __restrict__ Wc,
                              const float* __restrict__ cb, float* __restrict__ co) {
  int lane = threadIdx.x & 31;
  int m0   = (threadIdx.x >> 5) * 16;
  int n0   = blockIdx.x * 16;
  v8f acc = {};
  const __bf16* A  = cat + (size_t)m0 * (2 * H_);
  const __bf16* Bm = Wc  + (size_t)n0 * (2 * H_);
  for (int k0 = 0; k0 < 2 * H_; k0 += 32)
    acc = WMMA_BF16(ldA(A, 2 * H_, k0, lane), ldB(Bm, 2 * H_, k0, lane), acc);
  int col  = lane & 15;
  int half = (lane & 16) ? 8 : 0;
  float bias = cb[n0 + col];
#pragma unroll
  for (int r = 0; r < 8; r++)
    co[(size_t)(m0 + half + r) * H_ + n0 + col] = tanhf(acc[r] + bias);
}

// ---------------------------------------------------------------------------
// output = co @ out_w^T + out_b  (128 x 50257), f32 WMMA 16x16x4 directly on
// fp32 weights — bandwidth-bound on out_w (206 MB @ 23.3 TB/s ≈ 8.8 us), so
// no bf16 cast pass; prefetch the weight stream 1KB ahead.
// ---------------------------------------------------------------------------
__global__ void k_out_gemm(const float* __restrict__ co, const float* __restrict__ Wo,
                           const float* __restrict__ ob, float* __restrict__ out) {
  int lane = threadIdx.x & 31;
  int m0   = (threadIdx.x >> 5) * 16;
  int n0   = blockIdx.x * 16;
  int col  = lane & 15;
  int kh   = (lane & 16) ? 2 : 0;
  int nrow = n0 + col; if (nrow > V_ - 1) nrow = V_ - 1;   // clamp tail reads
  const float* Ap = co + (size_t)(m0 + col) * H_ + kh;
  const float* Bp = Wo + (size_t)nrow * H_ + kh;
  v8f acc = {};
  for (int kc = 0; kc < H_; kc += 64) {
    __builtin_prefetch(Bp + kc + 256, 0, 1);   // global_prefetch_b8, 1KB ahead
#pragma unroll
    for (int k0 = kc; k0 < kc + 64; k0 += 4) {
      v2f a = *(const v2f*)(Ap + k0);
      v2f b = *(const v2f*)(Bp + k0);
      acc = __builtin_amdgcn_wmma_f32_16x16x4_f32(false, a, false, b, (short)0, acc, false, false);
    }
  }
  int half = (lane & 16) ? 8 : 0;
  int n = n0 + col;
  if (n < V_) {
    float bias = ob[n];
#pragma unroll
    for (int r = 0; r < 8; r++)
      out[(size_t)(m0 + half + r) * V_ + n] = acc[r] + bias;
  }
}

// ---------------------------------------------------------------------------
extern "C" void kernel_launch(void* const* d_in, const int* in_sizes, int n_in,
                              void* d_out, int out_size, void* d_ws, size_t ws_size,
                              hipStream_t stream) {
  const int*   input_seq = (const int*)d_in[0];
  const float* h0       = (const float*)d_in[1];
  const float* c0       = (const float*)d_in[2];
  const float* enc_in   = (const float*)d_in[3];
  const float* emb      = (const float*)d_in[4];
  const float* W_ih     = (const float*)d_in[5];
  const float* W_hh     = (const float*)d_in[6];
  const float* b_ih     = (const float*)d_in[7];
  const float* b_hh     = (const float*)d_in[8];
  const float* W1_w     = (const float*)d_in[9];
  const float* W1_b     = (const float*)d_in[10];
  const float* vvec     = (const float*)d_in[11];
  const float* concat_w = (const float*)d_in[12];
  const float* concat_b = (const float*)d_in[13];
  const float* out_w    = (const float*)d_in[14];
  const float* out_b    = (const float*)d_in[15];

  float* out   = (float*)d_out;                   // (B,V)
  float* h_out = out + (size_t)B_ * V_;           // (1,B,H)
  float* c_out = h_out + (size_t)B_ * H_;         // (1,B,H)

  char* ws = (char*)d_ws;
  size_t off = 0;
  auto alloc = [&](size_t bytes) -> char* {
    char* p = ws + off;
    off = (off + bytes + 255) & ~(size_t)255;
    return p;
  };
  __bf16* xbf    = (__bf16*)alloc((size_t)B_ * W_ * 2);
  __bf16* hbf    = (__bf16*)alloc((size_t)B_ * H_ * 2);
  __bf16* Wihbf  = (__bf16*)alloc((size_t)G_ * W_ * 2);
  __bf16* Whhbf  = (__bf16*)alloc((size_t)G_ * H_ * 2);
  __bf16* Whbf   = (__bf16*)alloc((size_t)H_ * H_ * 2);
  __bf16* Webf   = (__bf16*)alloc((size_t)H_ * H_ * 2);
  __bf16* Wcbf   = (__bf16*)alloc((size_t)H_ * 2 * H_ * 2);
  __bf16* encbf  = (__bf16*)alloc((size_t)B_ * L_ * H_ * 2);   // 64 MB
  float*  gates  = (float*)alloc((size_t)B_ * G_ * 4);
  float*  hWh    = (float*)alloc((size_t)B_ * H_ * 4);
  float*  scores = (float*)alloc((size_t)B_ * L_ * 4);
  __bf16* catbf  = (__bf16*)alloc((size_t)B_ * 2 * H_ * 2);
  float*  co     = (float*)alloc((size_t)B_ * H_ * 4);

  // --- prep / casts ---
  k_embed<<<(B_ * W_ + 255) / 256, 256, 0, stream>>>(input_seq, emb, xbf);
  k_cast_bf16<<<(B_ * H_ + 255) / 256, 256, 0, stream>>>(h0, hbf, B_ * H_);
  k_cast_bf16<<<(G_ * W_ + 255) / 256, 256, 0, stream>>>(W_ih, Wihbf, G_ * W_);
  k_cast_bf16<<<(G_ * H_ + 255) / 256, 256, 0, stream>>>(W_hh, Whhbf, G_ * H_);
  k_cast_bf16<<<(H_ * 2 * H_ + 255) / 256, 256, 0, stream>>>(concat_w, Wcbf, H_ * 2 * H_);
  k_split_w1<<<(H_ * 2 * H_ + 255) / 256, 256, 0, stream>>>(W1_w, Whbf, Webf);
  k_enc_cast<<<(int)(((long long)L_ * B_ * H_ + 255) / 256), 256, 0, stream>>>(enc_in, encbf);
  k_zero_f32<<<(B_ * L_ + 255) / 256, 256, 0, stream>>>(scores, B_ * L_);

  // --- LSTM cell ---
  k_gates_gemm<<<G_ / 16, 256, 0, stream>>>(xbf, hbf, Wihbf, Whhbf, b_ih, b_hh, gates);
  k_lstm_pw<<<(B_ * H_ + 255) / 256, 256, 0, stream>>>(gates, c0, h_out, c_out, hbf, catbf);

  // --- attention ---
  k_hwh_gemm<<<H_ / 16, 256, 0, stream>>>(hbf, Whbf, W1_b, hWh);
  k_energy<<<dim3((B_ * L_ / 16) / 8, H_ / 64), 256, 0, stream>>>(encbf, Webf, hWh, vvec, scores);
  k_softmax_ctx<<<B_, 256, 0, stream>>>(scores, encbf, catbf);

  // --- heads ---
  k_concat_gemm<<<H_ / 16, 256, 0, stream>>>(catbf, Wcbf, concat_b, co);
  k_out_gemm<<<(V_ + 15) / 16, 256, 0, stream>>>(co, out_w, out_b, out);
}
